// EdgeSelector_75857712382001
// MI455X (gfx1250) — compile-verified, matrix operand order checked by reading, not run on hardware
//
#include <hip/hip_runtime.h>
#include <hip/hip_bf16.h>

#define DD 128
#define HID 128
#define NODES_PER_G 500
#define CAND_PER_G 5000

typedef __attribute__((ext_vector_type(16))) __bf16       v16bf;
typedef __attribute__((ext_vector_type(8)))  float        v8f;
typedef __attribute__((ext_vector_type(8)))  unsigned int v8u;
typedef __attribute__((ext_vector_type(4)))  unsigned int v4u;

static __device__ __forceinline__ unsigned short f2bf(float f) {
    unsigned u = __float_as_uint(f);
    u = (u + 0x7FFFu + ((u >> 16) & 1u)) >> 16;   // round-to-nearest-even
    return (unsigned short)u;
}

static __device__ __forceinline__ v8f wmma_bf16(v8u a, v8u b, v8f c) {
    return __builtin_amdgcn_wmma_f32_16x16x32_bf16(
        false, __builtin_bit_cast(v16bf, a),
        false, __builtin_bit_cast(v16bf, b),
        (short)0, c, false, false);
}

// A-operand (16x32 bf16) per-lane data: two contiguous 16B chunks at elem off eb, eb+16
static __device__ __forceinline__ v8u load_a(const unsigned short* p) {
    v4u lo = *(const v4u*)p;
    v4u hi = *(const v4u*)(p + 16);
    v8u r = {lo[0], lo[1], lo[2], lo[3], hi[0], hi[1], hi[2], hi[3]};
    return r;
}

static __device__ __forceinline__ v8u load_b(const unsigned int* q) {
    v4u b0 = ((const v4u*)q)[0];
    v4u b1 = ((const v4u*)q)[1];
    v8u r = {b0[0], b0[1], b0[2], b0[3], b1[0], b1[1], b1[2], b1[3]};
    return r;
}

static __device__ __forceinline__ v8f vzero8() {
    return (v8f){0.f,0.f,0.f,0.f,0.f,0.f,0.f,0.f};
}

// ---------------- f32 -> bf16 bulk convert ----------------
__global__ void cvt_bf16_kernel(const float* __restrict__ src,
                                unsigned short* __restrict__ dst, int n4) {
    int i = blockIdx.x * blockDim.x + threadIdx.x;
    if (i >= n4) return;
    float4 v = ((const float4*)src)[i];
    unsigned long long packed =
        (unsigned long long)f2bf(v.x)
      | ((unsigned long long)f2bf(v.y) << 16)
      | ((unsigned long long)f2bf(v.z) << 32)
      | ((unsigned long long)f2bf(v.w) << 48);
    ((unsigned long long*)dst)[i] = packed;
}

// ---------------- pack [K, ncols] f32 weight into WMMA B-operand layout ----------
// dst u32 index: ((nb*KS + ks)*32 + lane)*8 + s
__global__ void pack_weight_kernel(const float* __restrict__ src,
                                   unsigned int* __restrict__ dst,
                                   int KS, int NB, int K, int ncols) {
    int t = blockIdx.x * blockDim.x + threadIdx.x;
    int total = NB * KS * 32 * 8;
    if (t >= total) return;
    int s    = t & 7;
    int lane = (t >> 3) & 31;
    int ks   = (t >> 8) % KS;
    int nb   = t / (256 * KS);
    int n    = nb * 16 + (lane & 15);
    int kb   = ks * 32 + ((s < 4) ? 0 : 16) + ((lane < 16) ? 0 : 8) + 2 * (s & 3);
    float v0 = 0.f, v1 = 0.f;
    if (n < ncols) {
        if (kb     < K) v0 = src[(size_t)kb * ncols + n];
        if (kb + 1 < K) v1 = src[(size_t)(kb + 1) * ncols + n];
    }
    dst[t] = (unsigned)f2bf(v0) | ((unsigned)f2bf(v1) << 16);
}

// ---------------- encoder: x = nf @ W_enc + b_enc (bf16 in, bf16 out, f32 acc) ----
__global__ __launch_bounds__(128, 1)
void encoder_kernel(const unsigned short* __restrict__ nfb,
                    const unsigned int* __restrict__ wencp,
                    const float* __restrict__ benc,
                    unsigned short* __restrict__ xb,
                    int nrows, int ntiles) {
    int lane = threadIdx.x & 31;
    int ln   = lane & 15;
    int half = lane >> 4;
    int wid  = (blockIdx.x * blockDim.x + threadIdx.x) >> 5;
    int nw   = (gridDim.x * blockDim.x) >> 5;

    // loop-invariant: full W_enc B-operands resident in VGPRs (32 x v8u)
    v8u w0r[32];
    #pragma unroll
    for (int i = 0; i < 32; i++) w0r[i] = load_b(wencp + (i * 32 + lane) * 8);
    float biasE[8];
    #pragma unroll
    for (int nb = 0; nb < 8; nb++) biasE[nb] = benc[nb * 16 + ln];

    for (int t = wid; t < ntiles; t += nw) {
        int rid = t * 16 + ln;
        if (rid >= nrows) rid = nrows - 1;
        const unsigned short* rowp = nfb + (size_t)rid * DD;
        v8u a[4];
        #pragma unroll
        for (int ks = 0; ks < 4; ks++)
            a[ks] = load_a(rowp + ks * 32 + half * 8);

        v8f acc[8];
        #pragma unroll
        for (int nb = 0; nb < 8; nb++) acc[nb] = vzero8();
        #pragma unroll
        for (int ks = 0; ks < 4; ks++)          // ks outer -> 8 independent chains
            #pragma unroll
            for (int nb = 0; nb < 8; nb++)
                acc[nb] = wmma_bf16(a[ks], w0r[nb * 4 + ks], acc[nb]);

        bool fullTile = (t * 16 + 15) < nrows;  // wave-uniform
        #pragma unroll
        for (int nb = 0; nb < 8; nb++) {
            int n = nb * 16 + ln;
            if (fullTile) {
                #pragma unroll
                for (int j = 0; j < 8; j++) {
                    int r = t * 16 + j + half * 8;
                    xb[(size_t)r * DD + n] = f2bf(acc[nb][j] + biasE[nb]);
                }
            } else {
                #pragma unroll
                for (int j = 0; j < 8; j++) {
                    int r = t * 16 + j + half * 8;
                    if (r < nrows) xb[(size_t)r * DD + n] = f2bf(acc[nb][j] + biasE[nb]);
                }
            }
        }
    }
}

// ---------------- fused 2-layer edge MLP (both heads) ----------------
// W1 staged in LDS (packed B-operand layout) so waves stay small (multi-wave/SIMD
// occupancy hides the gathered-A L2 latency). One 16-row tile per wave.
// mode 0: select  (eidx=[EC,2] relative, +g*500 offset) ; mode 1: delete (eidx=[2,E])
__global__ __launch_bounds__(128)
void edge_mlp_kernel(const unsigned short* __restrict__ xb,
                     const unsigned int* __restrict__ w1p,
                     const float* __restrict__ b1,
                     const unsigned int* __restrict__ w2p,
                     const float* __restrict__ b2,
                     const int* __restrict__ eidx,
                     int mode, int nrows, int ntiles,
                     float* __restrict__ out) {
    __shared__ __align__(16) unsigned int   w1s[8 * 8 * 32 * 8];    // 64KB packed W1
    __shared__ __align__(16) unsigned short hsm[4 * 16 * HID];      // 16KB h transpose

    int tid  = threadIdx.x;
    int lane = tid & 31;
    int ln   = lane & 15;
    int half = lane >> 4;
    int wid  = (blockIdx.x * blockDim.x + tid) >> 5;
    int nw   = (gridDim.x * blockDim.x) >> 5;
    unsigned short* hw = hsm + (tid >> 5) * (16 * HID);

    // cooperative stage of packed W1 into LDS (16384 u32)
    for (int i = tid; i < 16384 / 4; i += blockDim.x)
        ((v4u*)w1s)[i] = ((const v4u*)w1p)[i];

    // small invariants in registers
    v8u b2op[4];
    #pragma unroll
    for (int ks = 0; ks < 4; ks++) b2op[ks] = load_b(w2p + (ks * 32 + lane) * 8);
    float bias1[8];
    #pragma unroll
    for (int nb = 0; nb < 8; nb++) bias1[nb] = b1[nb * 16 + ln];
    float bias2 = (ln < 5) ? b2[ln] : 0.f;

    __syncthreads();

    for (int t = wid; t < ntiles; t += nw) {
        int rid = t * 16 + ln;
        if (rid >= nrows) rid = nrows - 1;
        int i0, i1;
        if (mode == 0) {
            int off = (rid / CAND_PER_G) * NODES_PER_G;
            i0 = eidx[rid * 2 + 0] + off;
            i1 = eidx[rid * 2 + 1] + off;
        } else {
            i0 = eidx[rid];
            i1 = eidx[nrows + rid];
        }
        const unsigned short* r0 = xb + (size_t)i0 * DD;
        const unsigned short* r1 = xb + (size_t)i1 * DD;

        // gathered A operands (concat row = [x[i0] | x[i1]], K=256): 16 x b128
        v8u a[8];
        #pragma unroll
        for (int ks = 0; ks < 8; ks++) {
            const unsigned short* rp = (ks < 4) ? r0 : r1;
            a[ks] = load_a(rp + (ks & 3) * 32 + half * 8);
        }

        // layer 1: [16x256]@[256x128]; B from LDS; 8 independent chains
        v8f acc[8];
        #pragma unroll
        for (int nb = 0; nb < 8; nb++) acc[nb] = vzero8();
        #pragma unroll
        for (int ks = 0; ks < 8; ks++) {
            #pragma unroll
            for (int nb = 0; nb < 8; nb++)
                acc[nb] = wmma_bf16(a[ks], load_b(w1s + ((nb * 8 + ks) * 32 + lane) * 8),
                                    acc[nb]);
        }

        // bias + ReLU + transpose C-layout -> A-layout via LDS (wave-private)
        #pragma unroll
        for (int nb = 0; nb < 8; nb++) {
            int n = nb * 16 + ln;
            #pragma unroll
            for (int j = 0; j < 8; j++) {
                int m = j + half * 8;
                float h = acc[nb][j] + bias1[nb];
                h = h > 0.f ? h : 0.f;
                hw[m * HID + n] = f2bf(h);
            }
        }

        // layer 2: [16x128] @ [128x16(pad of 5)]
        v8f o = vzero8();
        #pragma unroll
        for (int ks = 0; ks < 4; ks++)
            o = wmma_bf16(load_a(hw + ln * HID + ks * 32 + half * 8), b2op[ks], o);

        bool fullTile = (t * 16 + 15) < nrows;   // wave-uniform
        if (ln < 5) {
            if (fullTile) {
                #pragma unroll
                for (int j = 0; j < 8; j++) {
                    int m = j + half * 8;
                    out[(size_t)(t * 16 + m) * 5 + ln] = o[j] + bias2;
                }
            } else {
                #pragma unroll
                for (int j = 0; j < 8; j++) {
                    int m = j + half * 8;
                    int r = t * 16 + m;
                    if (r < nrows) out[(size_t)r * 5 + ln] = o[j] + bias2;
                }
            }
        }
    }
}

// ---------------- edge_candidate_idx output ----------------
__global__ void cand_idx_kernel(const int* __restrict__ ec,
                                float* __restrict__ out, int n2) {
    int i = blockIdx.x * blockDim.x + threadIdx.x;
    if (i >= n2) return;
    int r = i >> 1;
    int off = (r / CAND_PER_G) * NODES_PER_G;
    out[i] = (float)(ec[i] + off);
}

extern "C" void kernel_launch(void* const* d_in, const int* in_sizes, int n_in,
                              void* d_out, int out_size, void* d_ws, size_t ws_size,
                              hipStream_t stream) {
    const float* node_feat = (const float*)d_in[0];
    const float* W_enc = (const float*)d_in[1];
    const float* b_enc = (const float*)d_in[2];
    const float* W1a   = (const float*)d_in[3];
    const float* b1a   = (const float*)d_in[4];
    const float* W2a   = (const float*)d_in[5];
    const float* b2a   = (const float*)d_in[6];
    const float* W1d   = (const float*)d_in[7];
    const float* b1d   = (const float*)d_in[8];
    const float* W2d   = (const float*)d_in[9];
    const float* b2d   = (const float*)d_in[10];
    const int* edge_index     = (const int*)d_in[11];
    const int* edge_candidate = (const int*)d_in[12];

    const int N  = in_sizes[0] / DD;    // 50000
    const int E  = in_sizes[11] / 2;    // 1,000,000
    const int EC = in_sizes[12] / 2;    // 500,000

    // workspace carve-up
    char* ws = (char*)d_ws;
    unsigned short* nfb = (unsigned short*)ws; ws += (size_t)N * DD * 2;
    unsigned short* xb  = (unsigned short*)ws; ws += (size_t)N * DD * 2;
    unsigned int* wencp = (unsigned int*)ws;   ws += 8 * 4 * 32 * 8 * 4;
    unsigned int* w1ap  = (unsigned int*)ws;   ws += 8 * 8 * 32 * 8 * 4;
    unsigned int* w2ap  = (unsigned int*)ws;   ws += 1 * 4 * 32 * 8 * 4;
    unsigned int* w1dp  = (unsigned int*)ws;   ws += 8 * 8 * 32 * 8 * 4;
    unsigned int* w2dp  = (unsigned int*)ws;   ws += 1 * 4 * 32 * 8 * 4;

    float* outS = (float*)d_out;                 // [EC,5]
    float* outD = outS + (size_t)EC * 5;         // [E,5]
    float* outI = outD + (size_t)E * 5;          // [EC,2] indices as f32

    {   // 1) node_feat -> bf16
        int n4 = (N * DD) / 4;
        cvt_bf16_kernel<<<(n4 + 255) / 256, 256, 0, stream>>>(node_feat, nfb, n4);
    }
    // 2) pack weights into WMMA B-operand layouts
    pack_weight_kernel<<<(8*4*32*8 + 255)/256, 256, 0, stream>>>(W_enc, wencp, 4, 8, 128, 128);
    pack_weight_kernel<<<(8*8*32*8 + 255)/256, 256, 0, stream>>>(W1a,   w1ap,  8, 8, 256, 128);
    pack_weight_kernel<<<(1*4*32*8 + 255)/256, 256, 0, stream>>>(W2a,   w2ap,  4, 1, 128, 5);
    pack_weight_kernel<<<(8*8*32*8 + 255)/256, 256, 0, stream>>>(W1d,   w1dp,  8, 8, 256, 128);
    pack_weight_kernel<<<(1*4*32*8 + 255)/256, 256, 0, stream>>>(W2d,   w2dp,  4, 1, 128, 5);

    {   // 3) encoder GEMM
        int ntiles = (N + 15) / 16;
        int blocks = (ntiles + 3) / 4;
        encoder_kernel<<<blocks, 128, 0, stream>>>(nfb, wencp, b_enc, xb, N, ntiles);
    }
    {   // 4) select head
        int ntiles = (EC + 15) / 16;
        int blocks = (ntiles + 3) / 4;
        edge_mlp_kernel<<<blocks, 128, 0, stream>>>(xb, w1ap, b1a, w2ap, b2a,
                                                    edge_candidate, 0, EC, ntiles, outS);
    }
    {   // 5) delete head
        int ntiles = (E + 15) / 16;
        int blocks = (ntiles + 3) / 4;
        edge_mlp_kernel<<<blocks, 128, 0, stream>>>(xb, w1dp, b1d, w2dp, b2d,
                                                    edge_index, 1, E, ntiles, outD);
    }
    {   // 6) candidate global indices
        int n2 = EC * 2;
        cand_idx_kernel<<<(n2 + 255) / 256, 256, 0, stream>>>(edge_candidate, outI, n2);
    }
}
